// UMambaBlock_35957466202537
// MI455X (gfx1250) — matrix operand order repaired and checked by reading, not compiled, not run on hardware
//
#include <hip/hip_runtime.h>
#include <hip/hip_bf16.h>
#include <math.h>

typedef __bf16 bf16_t;
typedef __attribute__((ext_vector_type(16))) __bf16 v16bf;
typedef __attribute__((ext_vector_type(8)))  __bf16 v8bf;
typedef __attribute__((ext_vector_type(8)))  float  v8f;

#define IN_EPS 1e-5f

// Block = 128 threads (4 waves); block tile M=64, N=128, K=32.
// Each wave computes 16(M) x 128(N) = 8 WMMA accumulators.
// Double-buffered LDS: As[2] (4KB each) + Bt[2] (8KB each) = 24KB.

// ---------------------------------------------------------------------------
// Fragment loaders (wide DS loads).
// A (16x32 MxK), As row-major [m][32]: lane l: m = waveM + (l&15),
//   K = (l>>4)*16 + e -> one contiguous 32B run.
// B (32x16 KxN), Bt transposed [n][32]: lane l: n = nb + (l&15),
//   K = (l>>4)*16 + e -> one contiguous 32B run.
// D (16x16 f32): lane l: N = l&15, vgpr v: M = (l>>4)*8 + v.
// ---------------------------------------------------------------------------
__device__ __forceinline__ v16bf frag_a_lds(const bf16_t* As, int waveM, int lane) {
    const int m  = waveM + (lane & 15);
    const int kb = (lane >> 4) << 4;
    return *(const v16bf*)(As + m * 32 + kb);
}

__device__ __forceinline__ v16bf frag_b_lds(const bf16_t* Bt, int nb, int lane) {
    const int n  = nb + (lane & 15);
    const int kb = (lane >> 4) << 4;
    return *(const v16bf*)(Bt + n * 32 + kb);
}

__device__ __forceinline__ v8f v8f_zero() {
    v8f z = {0.f, 0.f, 0.f, 0.f, 0.f, 0.f, 0.f, 0.f};
    return z;
}

// ---------------------------------------------------------------------------
// 3x3 conv (pad 1, stride 1) as implicit GEMM with WMMA.
// K ordered as (r,s,c): k = (r*3+s)*Cin + c, so each 32-wide K-chunk shares
// one (r,s) -> constant-offset input loads, no per-k div/mod.
// M = Cout, N = B*H*W (=16384).  Output NCHW f32 + bias.
// ---------------------------------------------------------------------------
__global__ void __launch_bounds__(128)
conv3x3_wmma_kernel(const float* __restrict__ in, const float* __restrict__ w,
                    const float* __restrict__ bias, float* __restrict__ out,
                    int Cin, int Cout, int cinLog2) {
    __shared__ __align__(32) bf16_t As[2][64 * 32];
    __shared__ __align__(32) bf16_t Bt[2][128 * 32];
    const int tid  = threadIdx.x;
    const int lane = tid & 31;
    const int wid  = tid >> 5;
    const int n0   = blockIdx.x * 128;
    const int m0   = blockIdx.y * 64;
    const int K    = Cin * 9;
    const int nIter = K >> 5;

    v8f acc[8];
#pragma unroll
    for (int i = 0; i < 8; ++i) acc[i] = v8f_zero();

    const int arow = tid >> 1;            // A: m index 0..63 (16 halves per thread)
    const int akq  = (tid & 1) << 4;      // 0 or 16
    const int brow = tid;                 // B: n index 0..127 (32 halves per thread)
    const int bN   = n0 + brow;
    const int bb   = bN >> 12, bpix = bN & 4095;
    const int by   = bpix >> 6, bx = bpix & 63;

    float aF[16];
    float bF[32];

#define CONV_FETCH(K0)                                                          \
    {                                                                           \
        const int c0 = (K0) & (Cin - 1);                                        \
        const int rs = (K0) >> cinLog2;                                         \
        const int r  = rs / 3, s = rs - r * 3;                                  \
        const float* wp = w + (size_t)(m0 + arow) * K + (size_t)(c0 + akq) * 9 + rs; \
        _Pragma("unroll")                                                       \
        for (int j = 0; j < 16; ++j) aF[j] = wp[j * 9];                         \
        const int iy = by + r - 1, ix = bx + s - 1;                             \
        const bool inb = ((unsigned)iy < 64u) && ((unsigned)ix < 64u);          \
        const float* ip = in + (size_t)(((bb * Cin + c0) * 64 + iy) * 64 + ix); \
        _Pragma("unroll")                                                       \
        for (int kk = 0; kk < 32; ++kk) bF[kk] = inb ? ip[(size_t)kk * 4096] : 0.f; \
    }

#define CONV_STORE(BUF)                                                         \
    {                                                                           \
        v16bf pk;                                                               \
        _Pragma("unroll")                                                       \
        for (int j = 0; j < 16; ++j) pk[j] = (bf16_t)aF[j];                     \
        *(v16bf*)(As[BUF] + arow * 32 + akq) = pk;                              \
        v16bf p0, p1;                                                           \
        _Pragma("unroll")                                                       \
        for (int kk = 0; kk < 16; ++kk) { p0[kk] = (bf16_t)bF[kk]; p1[kk] = (bf16_t)bF[kk + 16]; } \
        *(v16bf*)(Bt[BUF] + brow * 32)      = p0;                               \
        *(v16bf*)(Bt[BUF] + brow * 32 + 16) = p1;                               \
    }

    CONV_FETCH(0)
    CONV_STORE(0)
    __syncthreads();

    for (int it = 0; it < nIter; ++it) {
        const int cur = it & 1;
        const bool more = (it + 1) < nIter;
        if (more) CONV_FETCH((it + 1) << 5)

        const v16bf a = frag_a_lds(As[cur], wid << 4, lane);
        v16bf bfr[8];
#pragma unroll
        for (int t = 0; t < 8; ++t) bfr[t] = frag_b_lds(Bt[cur], t * 16, lane);
#pragma unroll
        for (int t = 0; t < 8; ++t)
            acc[t] = __builtin_amdgcn_wmma_f32_16x16x32_bf16(
                false, a, false, bfr[t], (short)0, acc[t], false, false);

        if (more) CONV_STORE(cur ^ 1)
        __syncthreads();
    }

    // Epilogue: bias + scatter NCHW
#pragma unroll
    for (int t = 0; t < 8; ++t) {
#pragma unroll
        for (int v = 0; v < 8; ++v) {
            const int m = m0 + (wid << 4) + ((lane >> 4) << 3) + v;
            const int n = n0 + t * 16 + (lane & 15);
            const int b = n >> 12, pix = n & 4095;
            out[(size_t)(b * Cout + m) * 4096 + pix] = acc[t][v] + bias[m];
        }
    }
#undef CONV_FETCH
#undef CONV_STORE
}

// ---------------------------------------------------------------------------
// Y(N,M) = X(N,K) @ W(K,M) with WMMA.  Ragged M (x_proj M=40) and ragged K
// (dt_proj K=8) via zero-padded tiles.  out_mode: 0 = Y[n*ldy+m],
// 1 = NCHW scatter out[(b*M+m)*4096 + pix], n = b*4096 + pix.
// ---------------------------------------------------------------------------
__global__ void __launch_bounds__(128)
gemm_tn_wmma_kernel(const float* __restrict__ X, int ldx, int Kact,
                    const float* __restrict__ Wm, int Mcols,
                    float* __restrict__ Y, int ldy, int out_mode) {
    __shared__ __align__(32) bf16_t As[2][64 * 32];
    __shared__ __align__(32) bf16_t Bt[2][128 * 32];
    const int tid  = threadIdx.x;
    const int lane = tid & 31;
    const int wid  = tid >> 5;
    const int n0   = blockIdx.x * 128;
    const int m0   = blockIdx.y * 64;
    const int Kpad = (Kact + 31) & ~31;
    const int nIter = Kpad >> 5;

    v8f acc[8];
#pragma unroll
    for (int i = 0; i < 8; ++i) acc[i] = v8f_zero();

    const int arow = tid >> 1;            // m index 0..63
    const int akq  = (tid & 1) << 4;      // 0 or 16
    const int brow = tid;                 // n index 0..127
    const int am   = m0 + arow;

    float aF[16];
    float bF[32];

#define GEMM_FETCH(K0)                                                          \
    {                                                                           \
        _Pragma("unroll")                                                       \
        for (int j = 0; j < 16; ++j) {                                          \
            const int k = (K0) + akq + j;                                       \
            aF[j] = (k < Kact && am < Mcols) ? Wm[(size_t)k * Mcols + am] : 0.f; \
        }                                                                       \
        const float* xp = X + (size_t)(n0 + brow) * ldx + (K0);                 \
        if ((K0) + 31 < Kact) {                                                 \
            _Pragma("unroll")                                                   \
            for (int kk = 0; kk < 32; ++kk) bF[kk] = xp[kk];                    \
        } else {                                                                \
            _Pragma("unroll")                                                   \
            for (int kk = 0; kk < 32; ++kk)                                     \
                bF[kk] = ((K0) + kk < Kact) ? xp[kk] : 0.f;                     \
        }                                                                       \
    }

#define GEMM_STORE(BUF)                                                         \
    {                                                                           \
        v16bf pk;                                                               \
        _Pragma("unroll")                                                       \
        for (int j = 0; j < 16; ++j) pk[j] = (bf16_t)aF[j];                     \
        *(v16bf*)(As[BUF] + arow * 32 + akq) = pk;                              \
        v16bf p0, p1;                                                           \
        _Pragma("unroll")                                                       \
        for (int kk = 0; kk < 16; ++kk) { p0[kk] = (bf16_t)bF[kk]; p1[kk] = (bf16_t)bF[kk + 16]; } \
        *(v16bf*)(Bt[BUF] + brow * 32)      = p0;                               \
        *(v16bf*)(Bt[BUF] + brow * 32 + 16) = p1;                               \
    }

    GEMM_FETCH(0)
    GEMM_STORE(0)
    __syncthreads();

    for (int it = 0; it < nIter; ++it) {
        const int cur = it & 1;
        const bool more = (it + 1) < nIter;
        if (more) GEMM_FETCH((it + 1) << 5)

        const v16bf a = frag_a_lds(As[cur], wid << 4, lane);
        v16bf bfr[8];
#pragma unroll
        for (int t = 0; t < 8; ++t) bfr[t] = frag_b_lds(Bt[cur], t * 16, lane);
#pragma unroll
        for (int t = 0; t < 8; ++t)
            acc[t] = __builtin_amdgcn_wmma_f32_16x16x32_bf16(
                false, a, false, bfr[t], (short)0, acc[t], false, false);

        if (more) GEMM_STORE(cur ^ 1)
        __syncthreads();
    }

#pragma unroll
    for (int t = 0; t < 8; ++t) {
#pragma unroll
        for (int v = 0; v < 8; ++v) {
            const int m = m0 + (wid << 4) + ((lane >> 4) << 3) + v;
            const int n = n0 + t * 16 + (lane & 15);
            if (m < Mcols) {
                if (out_mode == 0) {
                    Y[(size_t)n * ldy + m] = acc[t][v];
                } else {
                    const int b = n >> 12, pix = n & 4095;
                    Y[(size_t)(b * Mcols + m) * 4096 + pix] = acc[t][v];
                }
            }
        }
    }
#undef GEMM_FETCH
#undef GEMM_STORE
}

// ---------------------------------------------------------------------------
// InstanceNorm + LeakyReLU (in-place):  h = inorm(h) + leaky(h) (+ resid)
// ---------------------------------------------------------------------------
__global__ void __launch_bounds__(256)
inorm_lrelu_kernel(float* __restrict__ h, const float* __restrict__ resid) {
    __shared__ float rs[256], rq[256];
    const int bc = blockIdx.x;
    float* p = h + (size_t)bc * 4096;
    float s = 0.f, q = 0.f;
    for (int i = threadIdx.x; i < 4096; i += 256) {
        const float v = p[i];
        s += v; q += v * v;
    }
    rs[threadIdx.x] = s; rq[threadIdx.x] = q;
    __syncthreads();
    for (int off = 128; off > 0; off >>= 1) {
        if ((int)threadIdx.x < off) {
            rs[threadIdx.x] += rs[threadIdx.x + off];
            rq[threadIdx.x] += rq[threadIdx.x + off];
        }
        __syncthreads();
    }
    const float mean = rs[0] * (1.f / 4096.f);
    const float var  = rq[0] * (1.f / 4096.f) - mean * mean;
    const float rstd = rsqrtf(var + IN_EPS);
    for (int i = threadIdx.x; i < 4096; i += 256) {
        const float v = p[i];
        float o = (v - mean) * rstd + (v > 0.f ? v : 0.01f * v);
        if (resid) o += resid[(size_t)bc * 4096 + i];
        p[i] = o;
    }
}

// ---------------------------------------------------------------------------
// LayerNorm over channels (C=128) per pixel; NCHW -> t (N=16384, 128)
// ---------------------------------------------------------------------------
__global__ void __launch_bounds__(128)
layernorm_kernel(const float* __restrict__ h, const float* __restrict__ g,
                 const float* __restrict__ be, float* __restrict__ t) {
    __shared__ float rs[128], rq[128];
    const int n = blockIdx.x;
    const int b = n >> 12, pix = n & 4095;
    const int c = threadIdx.x;
    const float v = h[(size_t)(b * 128 + c) * 4096 + pix];
    rs[c] = v; rq[c] = v * v;
    __syncthreads();
    for (int off = 64; off > 0; off >>= 1) {
        if (c < off) { rs[c] += rs[c + off]; rq[c] += rq[c + off]; }
        __syncthreads();
    }
    const float mean = rs[0] * (1.f / 128.f);
    const float var  = rq[0] * (1.f / 128.f) - mean * mean;
    const float rstd = rsqrtf(var + IN_EPS);
    t[(size_t)n * 128 + c] = (v - mean) * rstd * g[c] + be[c];
}

// ---------------------------------------------------------------------------
// Depthwise causal conv1d (k=4) + SiLU.  xr holds [xs | res] (stride 512).
// ---------------------------------------------------------------------------
__global__ void __launch_bounds__(256)
conv1d_silu_kernel(const float* __restrict__ xr, const float* __restrict__ w,
                   const float* __restrict__ bias, float* __restrict__ out) {
    const int n = blockIdx.x;        // 0..16383
    const int c = threadIdx.x;       // 0..255
    const int l = n & 4095, b = n >> 12;
    float acc = bias[c];
#pragma unroll
    for (int j = 0; j < 4; ++j) {
        const int ll = l - 3 + j;
        if (ll >= 0)
            acc += w[c * 4 + j] * xr[(size_t)((b << 12) + ll) * 512 + c];
    }
    out[(size_t)n * 256 + c] = acc / (1.f + __expf(-acc));
}

// ---------------------------------------------------------------------------
// Selective scan + gating.  One block per batch, one channel per thread.
// ---------------------------------------------------------------------------
__global__ void __launch_bounds__(256)
scan_gate_kernel(const float* __restrict__ xsa, const float* __restrict__ xdbl,
                 const float* __restrict__ dpre, const float* __restrict__ dt_b,
                 const float* __restrict__ A_log, const float* __restrict__ Dp,
                 const float* __restrict__ xr, float* __restrict__ yg) {
    __shared__ float sB[16], sC[16];
    const int b = blockIdx.x;
    const int d = threadIdx.x;
    float Ar[16], h[16];
#pragma unroll
    for (int j = 0; j < 16; ++j) { Ar[j] = -__expf(A_log[d * 16 + j]); h[j] = 0.f; }
    const float Dv = Dp[d];
    const float db = dt_b[d];
    for (int l = 0; l < 4096; ++l) {
        const int n = (b << 12) + l;
        if (d < 16)       sB[d]      = xdbl[(size_t)n * 40 + 8  + d];
        else if (d < 32)  sC[d - 16] = xdbl[(size_t)n * 40 + 24 + (d - 16)];
        __syncthreads();
        const float u = xsa[(size_t)n * 256 + d];
        const float xv = dpre[(size_t)n * 256 + d] + db;
        const float delta = (xv > 20.f) ? xv : log1pf(__expf(xv));
        const float du = delta * u;
        float y = 0.f;
#pragma unroll
        for (int j = 0; j < 16; ++j) {
            h[j] = __expf(delta * Ar[j]) * h[j] + du * sB[j];
            y += h[j] * sC[j];
        }
        y += u * Dv;
        const float r  = xr[(size_t)n * 512 + 256 + d];
        yg[(size_t)n * 256 + d] = y * (r / (1.f + __expf(-r)));
        __syncthreads();
    }
}

// ---------------------------------------------------------------------------
extern "C" void kernel_launch(void* const* d_in, const int* in_sizes, int n_in,
                              void* d_out, int out_size, void* d_ws, size_t ws_size,
                              hipStream_t stream) {
    (void)in_sizes; (void)n_in; (void)out_size; (void)ws_size;
    const float* x         = (const float*)d_in[0];
    const float* conv1_w   = (const float*)d_in[1];
    const float* conv1_b   = (const float*)d_in[2];
    const float* conv2_w   = (const float*)d_in[3];
    const float* conv2_b   = (const float*)d_in[4];
    const float* ln_g      = (const float*)d_in[5];
    const float* ln_b      = (const float*)d_in[6];
    const float* in_proj_w = (const float*)d_in[7];
    const float* conv1d_w  = (const float*)d_in[8];
    const float* conv1d_b  = (const float*)d_in[9];
    const float* x_proj_w  = (const float*)d_in[10];
    const float* dt_proj_w = (const float*)d_in[11];
    const float* dt_proj_b = (const float*)d_in[12];
    const float* A_log     = (const float*)d_in[13];
    const float* Dp        = (const float*)d_in[14];
    const float* out_pw    = (const float*)d_in[15];
    float* out = (float*)d_out;
    float* ws  = (float*)d_ws;

    // workspace layout (floats), with reuse
    float* h1   = ws;                 // 4*256*4096  = 4,194,304
    float* h2   = ws + 4194304;       // 4*128*4096  = 2,097,152
    float* t    = ws + 6291456;       // 16384*128   = 2,097,152
    float* xr   = ws + 8388608;       // 16384*512   = 8,388,608
    float* xsa  = ws + 16777216;      // 16384*256   = 4,194,304
    float* dpre = ws + 20971520;      // 16384*256   = 4,194,304
    float* xdbl = h2;                 // reuse: 16384*40 (h2 dead after layernorm)
    float* yg   = h1;                 // reuse: 16384*256 (h1 dead after conv2)

    // conv1 (128->256) + inorm/leaky
    conv3x3_wmma_kernel<<<dim3(128, 4), 128, 0, stream>>>(x, conv1_w, conv1_b, h1, 128, 256, 7);
    inorm_lrelu_kernel<<<dim3(4 * 256), 256, 0, stream>>>(h1, nullptr);
    // conv2 (256->128) + inorm/leaky + residual
    conv3x3_wmma_kernel<<<dim3(128, 2), 128, 0, stream>>>(h1, conv2_w, conv2_b, h2, 256, 128, 8);
    inorm_lrelu_kernel<<<dim3(4 * 128), 256, 0, stream>>>(h2, x);
    // layernorm -> t (16384,128)
    layernorm_kernel<<<dim3(16384), 128, 0, stream>>>(h2, ln_g, ln_b, t);
    // in_proj: (16384,128)@(128,512) -> xr
    gemm_tn_wmma_kernel<<<dim3(128, 8), 128, 0, stream>>>(t, 128, 128, in_proj_w, 512, xr, 512, 0);
    // depthwise conv1d + silu -> xsa (16384,256)
    conv1d_silu_kernel<<<dim3(16384), 256, 0, stream>>>(xr, conv1d_w, conv1d_b, xsa);
    // x_proj: (16384,256)@(256,40) -> xdbl
    gemm_tn_wmma_kernel<<<dim3(128, 1), 128, 0, stream>>>(xsa, 256, 256, x_proj_w, 40, xdbl, 40, 0);
    // dt_proj: (16384,8)@(8,256) -> dpre
    gemm_tn_wmma_kernel<<<dim3(128, 4), 128, 0, stream>>>(xdbl, 40, 8, dt_proj_w, 256, dpre, 256, 0);
    // selective scan + gating -> yg
    scan_gate_kernel<<<dim3(4), 256, 0, stream>>>(xsa, xdbl, dpre, dt_proj_b, A_log, Dp, xr, yg);
    // out_proj: (16384,256)@(256,128) -> out (NCHW)
    gemm_tn_wmma_kernel<<<dim3(128, 2), 128, 0, stream>>>(yg, 256, 256, out_pw, 128, out, 0, 1);
}
